// Net_34110630265424
// MI455X (gfx1250) — 16-bit and 8-bit matrix operand order repaired and checked by reading, compiled, not run on hardware
//
#include <hip/hip_runtime.h>
#include <hip/hip_bf16.h>
#include <stdint.h>

#define B_DIM 16384
#define D_IN  784
#define KPAD  800
#define H_DIM 3072
#define D_OUT 10
#define EPS   1e-5f

#define BM 128
#define BN 128
#define LSTR 80

typedef __attribute__((ext_vector_type(8)))  int    v8i;
typedef __attribute__((ext_vector_type(8)))  float  v8f;
typedef __attribute__((ext_vector_type(16))) __bf16 v16bf;

__device__ __forceinline__ void async_lds_b128(void* lds_ptr, const void* gaddr) {
  unsigned lds = (unsigned)(unsigned long long)lds_ptr;
  asm volatile("global_load_async_to_lds_b128 %0, %1, off"
               :: "v"(lds), "v"(gaddr) : "memory");
}
__device__ __forceinline__ void wait_async0() {
  asm volatile("s_wait_asynccnt 0x0" ::: "memory");
}

__global__ void prep_x_kernel(const float* __restrict__ x,
                              __bf16* __restrict__ xhi, __bf16* __restrict__ xlo) {
  const int m = blockIdx.x;
  const float* xr = x + (size_t)m * D_IN;
  __bf16* hr = xhi + (size_t)m * KPAD;
  __bf16* lr = xlo + (size_t)m * KPAD;
  for (int k = threadIdx.x; k < KPAD; k += blockDim.x) {
    float v = (k < D_IN) ? xr[k] : 0.0f;
    __bf16 hi = (__bf16)v;
    __bf16 lo = (__bf16)(v - (float)hi);
    hr[k] = hi; lr[k] = lo;
  }
}

__global__ void prep_w1_kernel(const float* __restrict__ w1, __bf16* __restrict__ w1s) {
  const int n = blockIdx.x;
  const float* wr = w1 + (size_t)n * D_IN;
  __bf16* sr = w1s + (size_t)n * KPAD;
  for (int k = threadIdx.x; k < KPAD; k += blockDim.x) {
    float v = 0.0f;
    if (k < D_IN) { float w = wr[k]; v = (w > 0.f) ? 1.f : ((w < 0.f) ? -1.f : 0.f); }
    sr[k] = (__bf16)v;
  }
}

__global__ void prep_wsign_kernel(const float* __restrict__ w, int8_t* __restrict__ ws) {
  const int idx = (blockIdx.x * blockDim.x + threadIdx.x) * 4;
  float4 v = *(const float4*)(w + idx);
  char4 s;
  s.x = (v.x > 0.f) ? 1 : ((v.x < 0.f) ? -1 : 0);
  s.y = (v.y > 0.f) ? 1 : ((v.y < 0.f) ? -1 : 0);
  s.z = (v.z > 0.f) ? 1 : ((v.z < 0.f) ? -1 : 0);
  s.w = (v.w > 0.f) ? 1 : ((v.w < 0.f) ? -1 : 0);
  *(char4*)(ws + idx) = s;
}

__global__ __launch_bounds__(256) void gemm_bf16_kernel(
    const __bf16* __restrict__ Ahi, const __bf16* __restrict__ Alo,
    const __bf16* __restrict__ Bw,  const float* __restrict__ bias,
    float* __restrict__ C) {
  __shared__ char lAh[2][BM * LSTR];
  __shared__ char lAl[2][BM * LSTR];
  __shared__ char lB [2][BN * LSTR];

  const int tid   = threadIdx.x;
  const int lane  = tid & 31;
  const int wave  = tid >> 5;
  const int wm    = wave & 1;
  const int wn    = wave >> 1;
  const int rowBlk = blockIdx.x * BM;
  const int colBlk = blockIdx.y * BN;
  const int lr = tid >> 2;
  const int lc = (tid & 3) * 16;
  const int mrow  = lane & 15;
  const int lhalf = lane >> 4;

  const v8f zf = {0,0,0,0,0,0,0,0};
  v8f acc[4][2];
#pragma unroll
  for (int i = 0; i < 4; ++i)
#pragma unroll
    for (int j = 0; j < 2; ++j) acc[i][j] = zf;

#pragma unroll
  for (int it = 0; it < 2; ++it) {
    int r = lr + it * 64;
    async_lds_b128(&lAh[0][r*LSTR + lc], (const char*)(Ahi + (size_t)(rowBlk+r)*KPAD) + lc);
    async_lds_b128(&lAl[0][r*LSTR + lc], (const char*)(Alo + (size_t)(rowBlk+r)*KPAD) + lc);
    async_lds_b128(&lB [0][r*LSTR + lc], (const char*)(Bw  + (size_t)(colBlk+r)*KPAD) + lc);
  }
  wait_async0();
  __syncthreads();

  const int KSTEPS = KPAD / 32;
  for (int ks = 0; ks < KSTEPS; ++ks) {
    const int buf = ks & 1;
    if (ks + 1 < KSTEPS) {
      const int nb = buf ^ 1;
      const int k0 = (ks + 1) * 32;
#pragma unroll
      for (int it = 0; it < 2; ++it) {
        int r = lr + it * 64;
        async_lds_b128(&lAh[nb][r*LSTR + lc], (const char*)(Ahi + (size_t)(rowBlk+r)*KPAD + k0) + lc);
        async_lds_b128(&lAl[nb][r*LSTR + lc], (const char*)(Alo + (size_t)(rowBlk+r)*KPAD + k0) + lc);
        async_lds_b128(&lB [nb][r*LSTR + lc], (const char*)(Bw  + (size_t)(colBlk+r)*KPAD + k0) + lc);
      }
    }
    union HF { int4 q[2]; v16bf v; };
    HF bfr[2], ah[4], al[4];
#pragma unroll
    for (int ni = 0; ni < 2; ++ni) {
      int rB = wn*32 + ni*16 + mrow;
      bfr[ni].q[0] = *(const int4*)&lB[buf][rB*LSTR +      lhalf*16];
      bfr[ni].q[1] = *(const int4*)&lB[buf][rB*LSTR + 32 + lhalf*16];
    }
#pragma unroll
    for (int mi = 0; mi < 4; ++mi) {
      int rA = wm*64 + mi*16 + mrow;
      ah[mi].q[0] = *(const int4*)&lAh[buf][rA*LSTR +      lhalf*16];
      ah[mi].q[1] = *(const int4*)&lAh[buf][rA*LSTR + 32 + lhalf*16];
      al[mi].q[0] = *(const int4*)&lAl[buf][rA*LSTR +      lhalf*16];
      al[mi].q[1] = *(const int4*)&lAl[buf][rA*LSTR + 32 + lhalf*16];
    }
#pragma unroll
    for (int mi = 0; mi < 4; ++mi)
#pragma unroll
      for (int ni = 0; ni < 2; ++ni) {
        acc[mi][ni] = __builtin_amdgcn_wmma_f32_16x16x32_bf16(
            false, ah[mi].v, false, bfr[ni].v, (short)0, acc[mi][ni], false, false);
        acc[mi][ni] = __builtin_amdgcn_wmma_f32_16x16x32_bf16(
            false, al[mi].v, false, bfr[ni].v, (short)0, acc[mi][ni], false, false);
      }
    wait_async0();
    __syncthreads();
  }

#pragma unroll
  for (int mi = 0; mi < 4; ++mi)
#pragma unroll
    for (int ni = 0; ni < 2; ++ni) {
      int col = colBlk + wn*32 + ni*16 + mrow;
      float bv = bias[col];
#pragma unroll
      for (int j = 0; j < 8; ++j) {
        int row = rowBlk + wm*64 + mi*16 + lhalf*8 + j;
        C[(size_t)row * H_DIM + col] = acc[mi][ni][j] + bv;
      }
    }
}

__global__ __launch_bounds__(256) void gemm_iu8_kernel(
    const int8_t* __restrict__ A, const int8_t* __restrict__ W,
    const float* __restrict__ bias, float* __restrict__ C) {
  __shared__ char lA[2][BM * LSTR];
  __shared__ char lB[2][BN * LSTR];

  const int tid   = threadIdx.x;
  const int lane  = tid & 31;
  const int wave  = tid >> 5;
  const int wm    = wave & 1;
  const int wn    = wave >> 1;
  const int rowBlk = blockIdx.x * BM;
  const int colBlk = blockIdx.y * BN;
  const int lr = tid >> 2;
  const int lc = (tid & 3) * 16;
  const int mrow  = lane & 15;
  const int lhalf = lane >> 4;
  const int hoff  = lhalf * 8;

  const v8i zi = {0,0,0,0,0,0,0,0};
  v8i acc[4][2];
#pragma unroll
  for (int i = 0; i < 4; ++i)
#pragma unroll
    for (int j = 0; j < 2; ++j) acc[i][j] = zi;

#pragma unroll
  for (int it = 0; it < 2; ++it) {
    int r = lr + it * 64;
    async_lds_b128(&lA[0][r*LSTR + lc], A + (size_t)(rowBlk+r)*H_DIM + lc);
    async_lds_b128(&lB[0][r*LSTR + lc], W + (size_t)(colBlk+r)*H_DIM + lc);
  }
  wait_async0();
  __syncthreads();

  const int KSTEPS = H_DIM / 64;
  for (int ks = 0; ks < KSTEPS; ++ks) {
    const int buf = ks & 1;
    if (ks + 1 < KSTEPS) {
      const int nb = buf ^ 1;
      const int k0 = (ks + 1) * 64;
#pragma unroll
      for (int it = 0; it < 2; ++it) {
        int r = lr + it * 64;
        async_lds_b128(&lA[nb][r*LSTR + lc], A + (size_t)(rowBlk+r)*H_DIM + k0 + lc);
        async_lds_b128(&lB[nb][r*LSTR + lc], W + (size_t)(colBlk+r)*H_DIM + k0 + lc);
      }
    }
    union BU { int4 q[2]; v8i v; } bfr[2];
    union AU { unsigned long long l[4]; v8i v; } af[4];
#pragma unroll
    for (int ni = 0; ni < 2; ++ni) {
      int rB = wn*32 + ni*16 + mrow;
      bfr[ni].q[0] = *(const int4*)&lB[buf][rB*LSTR +      lhalf*16];
      bfr[ni].q[1] = *(const int4*)&lB[buf][rB*LSTR + 32 + lhalf*16];
    }
#pragma unroll
    for (int mi = 0; mi < 4; ++mi) {
      int rA = wm*64 + mi*16 + mrow;
#pragma unroll
      for (int g = 0; g < 4; ++g)
        af[mi].l[g] = *(const unsigned long long*)&lA[buf][rA*LSTR + (g>>1)*32 + lhalf*16 + (g&1)*8];
    }
#pragma unroll
    for (int mi = 0; mi < 4; ++mi)
#pragma unroll
      for (int ni = 0; ni < 2; ++ni)
        acc[mi][ni] = __builtin_amdgcn_wmma_i32_16x16x64_iu8(
            true, af[mi].v, true, bfr[ni].v, acc[mi][ni], false, false);
    wait_async0();
    __syncthreads();
  }

#pragma unroll
  for (int mi = 0; mi < 4; ++mi)
#pragma unroll
    for (int ni = 0; ni < 2; ++ni) {
      int col = colBlk + wn*32 + ni*16 + mrow;
      float bv = bias[col];
#pragma unroll
      for (int j = 0; j < 8; ++j) {
        int row = rowBlk + wm*64 + mi*16 + lhalf*8 + j;
        C[(size_t)row * H_DIM + col] = (float)acc[mi][ni][j] + bv;
      }
    }
}

__global__ void bn_partial_kernel(const float* __restrict__ h,
                                  float* __restrict__ psum, float* __restrict__ psq) {
  const int c  = blockIdx.x * blockDim.x + threadIdx.x;
  const int rb = blockIdx.y;
  const int rows = B_DIM / 64;
  const float* p = h + (size_t)rb * rows * H_DIM + c;
  float s = 0.f, s2 = 0.f;
  for (int r = 0; r < rows; ++r) {
    float v = p[(size_t)r * H_DIM];
    s += v; s2 += v * v;
  }
  psum[rb * H_DIM + c] = s;
  psq [rb * H_DIM + c] = s2;
}

__global__ void bn_finalize_kernel(const float* __restrict__ psum, const float* __restrict__ psq,
                                   const float* __restrict__ g, const float* __restrict__ be,
                                   float* __restrict__ scale, float* __restrict__ shift) {
  const int c = blockIdx.x * blockDim.x + threadIdx.x;
  float s = 0.f, s2 = 0.f;
  for (int rb = 0; rb < 64; ++rb) { s += psum[rb*H_DIM + c]; s2 += psq[rb*H_DIM + c]; }
  float mean = s * (1.0f / B_DIM);
  float var  = fmaxf(s2 * (1.0f / B_DIM) - mean * mean, 0.f);
  float sc = g[c] * rsqrtf(var + EPS);
  scale[c] = sc;
  shift[c] = be[c] - mean * sc;
}

__global__ void bn_apply_kernel(float* __restrict__ h,
                                const float* __restrict__ scale, const float* __restrict__ shift,
                                int8_t* __restrict__ asign) {
  const size_t idx = ((size_t)blockIdx.x * blockDim.x + threadIdx.x) * 4;
  const int c = (int)(idx % H_DIM);
  float4 v  = *(float4*)(h + idx);
  float4 sc = *(const float4*)(scale + c);
  float4 sh = *(const float4*)(shift + c);
  float4 y;
  y.x = fminf(fmaxf(v.x * sc.x + sh.x, -1.f), 1.f);
  y.y = fminf(fmaxf(v.y * sc.y + sh.y, -1.f), 1.f);
  y.z = fminf(fmaxf(v.z * sc.z + sh.z, -1.f), 1.f);
  y.w = fminf(fmaxf(v.w * sc.w + sh.w, -1.f), 1.f);
  *(float4*)(h + idx) = y;
  char4 s;
  s.x = (y.x > 0.f) ? 1 : ((y.x < 0.f) ? -1 : 0);
  s.y = (y.y > 0.f) ? 1 : ((y.y < 0.f) ? -1 : 0);
  s.z = (y.z > 0.f) ? 1 : ((y.z < 0.f) ? -1 : 0);
  s.w = (y.w > 0.f) ? 1 : ((y.w < 0.f) ? -1 : 0);
  *(char4*)(asign + idx) = s;
}

__global__ __launch_bounds__(256) void fc4_kernel(const float* __restrict__ h,
                                                  const float* __restrict__ w4,
                                                  const float* __restrict__ b4,
                                                  float* __restrict__ out) {
  const int gid  = blockIdx.x * blockDim.x + threadIdx.x;
  const int row  = gid >> 5;
  const int lane = gid & 31;
  const float* hr = h + (size_t)row * H_DIM;
  float acc[D_OUT];
#pragma unroll
  for (int o = 0; o < D_OUT; ++o) acc[o] = 0.f;
  for (int k = lane; k < H_DIM; k += 32) {
    float x = hr[k];
#pragma unroll
    for (int o = 0; o < D_OUT; ++o) acc[o] += x * w4[o * H_DIM + k];
  }
#pragma unroll
  for (int o = 0; o < D_OUT; ++o)
#pragma unroll
    for (int m = 16; m >= 1; m >>= 1) acc[o] += __shfl_xor(acc[o], m, 32);
  if (lane == 0) {
    float lg[D_OUT];
    float mx = -3.4e38f;
#pragma unroll
    for (int o = 0; o < D_OUT; ++o) { lg[o] = acc[o] + b4[o]; mx = fmaxf(mx, lg[o]); }
    float se = 0.f;
#pragma unroll
    for (int o = 0; o < D_OUT; ++o) se += expf(lg[o] - mx);
    float lse = mx + logf(se);
#pragma unroll
    for (int o = 0; o < D_OUT; ++o) out[(size_t)row * D_OUT + o] = lg[o] - lse;
  }
}

extern "C" void kernel_launch(void* const* d_in, const int* in_sizes, int n_in,
                              void* d_out, int out_size, void* d_ws, size_t ws_size,
                              hipStream_t stream) {
  (void)in_sizes; (void)n_in; (void)out_size; (void)ws_size;
  const float* x   = (const float*)d_in[0];
  const float* w1  = (const float*)d_in[1];
  const float* b1  = (const float*)d_in[2];
  const float* g1  = (const float*)d_in[3];
  const float* be1 = (const float*)d_in[4];
  const float* w2  = (const float*)d_in[5];
  const float* b2  = (const float*)d_in[6];
  const float* g2  = (const float*)d_in[7];
  const float* be2 = (const float*)d_in[8];
  const float* w3  = (const float*)d_in[9];
  const float* b3  = (const float*)d_in[10];
  const float* g3  = (const float*)d_in[11];
  const float* be3 = (const float*)d_in[12];
  const float* w4  = (const float*)d_in[13];
  const float* b4  = (const float*)d_in[14];
  float* out = (float*)d_out;

  char* ws = (char*)d_ws;
  size_t off = 0;
  auto carve = [&](size_t bytes) -> char* {
    char* p = ws + off;
    off = (off + bytes + 255) & ~(size_t)255;
    return p;
  };
  float*  h     = (float*) carve((size_t)B_DIM * H_DIM * 4);
  int8_t* asign = (int8_t*)carve((size_t)B_DIM * H_DIM);
  int8_t* wsign = (int8_t*)carve((size_t)H_DIM * H_DIM);
  __bf16* xhi   = (__bf16*)carve((size_t)B_DIM * KPAD * 2);
  __bf16* xlo   = (__bf16*)carve((size_t)B_DIM * KPAD * 2);
  __bf16* w1s   = (__bf16*)carve((size_t)H_DIM * KPAD * 2);
  float*  scale = (float*) carve(H_DIM * 4);
  float*  shift = (float*) carve(H_DIM * 4);
  float*  psum  = (float*) carve(64 * H_DIM * 4);
  float*  psq   = (float*) carve(64 * H_DIM * 4);

  const dim3 gemmGrid(B_DIM / BM, H_DIM / BN);
  const dim3 bnPartGrid(H_DIM / 256, 64);
  const int bnApplyBlocks = (B_DIM * (H_DIM / 4)) / 256;
  const int wsignBlocks   = (H_DIM * (H_DIM / 4)) / 256;

  prep_x_kernel <<<B_DIM, 256, 0, stream>>>(x, xhi, xlo);
  prep_w1_kernel<<<H_DIM, 256, 0, stream>>>(w1, w1s);
  gemm_bf16_kernel<<<gemmGrid, 256, 0, stream>>>(xhi, xlo, w1s, b1, h);
  bn_partial_kernel <<<bnPartGrid, 256, 0, stream>>>(h, psum, psq);
  bn_finalize_kernel<<<H_DIM / 256, 256, 0, stream>>>(psum, psq, g1, be1, scale, shift);
  bn_apply_kernel   <<<bnApplyBlocks, 256, 0, stream>>>(h, scale, shift, asign);

  prep_wsign_kernel<<<wsignBlocks, 256, 0, stream>>>(w2, wsign);
  gemm_iu8_kernel  <<<gemmGrid, 256, 0, stream>>>(asign, wsign, b2, h);
  bn_partial_kernel <<<bnPartGrid, 256, 0, stream>>>(h, psum, psq);
  bn_finalize_kernel<<<H_DIM / 256, 256, 0, stream>>>(psum, psq, g2, be2, scale, shift);
  bn_apply_kernel   <<<bnApplyBlocks, 256, 0, stream>>>(h, scale, shift, asign);

  prep_wsign_kernel<<<wsignBlocks, 256, 0, stream>>>(w3, wsign);
  gemm_iu8_kernel  <<<gemmGrid, 256, 0, stream>>>(asign, wsign, b3, h);
  bn_partial_kernel <<<bnPartGrid, 256, 0, stream>>>(h, psum, psq);
  bn_finalize_kernel<<<H_DIM / 256, 256, 0, stream>>>(psum, psq, g3, be3, scale, shift);
  bn_apply_kernel   <<<bnApplyBlocks, 256, 0, stream>>>(h, scale, shift, asign);

  fc4_kernel<<<(B_DIM * 32) / 256, 256, 0, stream>>>(h, w4, b4, out);
}
